// LSTMPPOPolicy_15590731284789
// MI455X (gfx1250) — matrix operand order repaired and checked by reading, not compiled, not run on hardware
//
#include <hip/hip_runtime.h>
#include <hip/hip_bf16.h>

typedef __attribute__((ext_vector_type(16))) _Float16 v16h;
typedef __attribute__((ext_vector_type(4)))  _Float16 v4h;
typedef __attribute__((ext_vector_type(8)))  float    v8f;

#define WMMA_F16(a, b, c) \
  __builtin_amdgcn_wmma_f32_16x16x32_f16(false, (a), false, (b), (short)0, (c), false, false)

// ---------------- problem constants ----------------
#define B_   256
#define T_   256
#define D_   128
#define E_   256
#define H_   256
#define A_   18
#define G4H  1024                  // 4*H
#define BT   (B_ * T_)             // 65536
#define NWG  16                    // workgroups in recurrent kernel

// ---------------- output layout (float elements) ----------------
constexpr size_t OUT_LOGITS = 0;
constexpr size_t OUT_VALUES = OUT_LOGITS + (size_t)BT * A_;
constexpr size_t OUT_HT     = OUT_VALUES + (size_t)BT;
constexpr size_t OUT_CT     = OUT_HT + (size_t)B_ * H_;
constexpr size_t OUT_AR     = OUT_CT + (size_t)B_ * H_;
constexpr size_t OUT_TAR    = OUT_AR + 1;
constexpr size_t OUT_I      = OUT_TAR + 1;
constexpr size_t GSZ        = (size_t)BT * H_;
constexpr size_t OUT_F      = OUT_I + GSZ;
constexpr size_t OUT_G      = OUT_F + GSZ;
constexpr size_t OUT_O      = OUT_G + GSZ;
constexpr size_t OUT_CS     = OUT_O + GSZ;
constexpr size_t OUT_HS     = OUT_CS + GSZ;

// ---------------- workspace layout (bytes) ----------------
constexpr size_t WS_CTR   = 0;                                     // int counter
constexpr size_t WS_H16   = 256;                                   // 2 * B*H f16
constexpr size_t WS_HS16  = WS_H16   + 2ull * B_ * H_ * 2;          // BT*H f16
constexpr size_t WS_ENC1  = WS_HS16  + (size_t)BT * H_ * 2;         // BT*E f16
constexpr size_t WS_ENC2  = WS_ENC1  + (size_t)BT * E_ * 2;         // BT*E f16
constexpr size_t WS_XG    = WS_ENC2  + (size_t)BT * E_ * 2;         // BT*4H f16 (time-major)
constexpr size_t WS_OBS16 = WS_XG    + (size_t)BT * G4H * 2;        // BT*D f16
constexpr size_t WS_W1    = WS_OBS16 + (size_t)BT * D_ * 2;         // E*D f16
constexpr size_t WS_W2    = WS_W1    + (size_t)E_ * D_ * 2;         // E*E f16
constexpr size_t WS_WIH   = WS_W2    + (size_t)E_ * E_ * 2;         // 4H*E f16
constexpr size_t WS_WHH   = WS_WIH   + (size_t)G4H * E_ * 2;        // 4H*H f16
constexpr size_t WS_WPAD  = WS_WHH   + (size_t)G4H * H_ * 2;        // 32*H f16 (actor+critic+zeros)

// ---------------- helpers ----------------
__device__ __forceinline__ float sigmoidf_(float x) {
  return 1.0f / (1.0f + __expf(-x));
}
__device__ __forceinline__ float tanhf_(float x) {
  return 2.0f / (1.0f + __expf(-2.0f * x)) - 1.0f;   // safe at +/-inf
}

// A fragment (16x32 f16): lanes 0-15 rows, K split by lane-half per ISA layout.
__device__ __forceinline__ v16h load_a_f16(const _Float16* __restrict__ p, int hi) {
  v16h a;
#pragma unroll
  for (int i = 0; i < 8; ++i) {
    a[i]     = p[hi * 8 + i];          // K = k0 + hi*8 + i
    a[8 + i] = p[16 + hi * 8 + i];     // K = k0 + 16 + hi*8 + i
  }
  return a;
}
// B fragment (32x16): lane = column n; B[k][n] = W[n][k], W row-major (N x K) f16.
__device__ __forceinline__ v16h load_b_f16(const _Float16* __restrict__ p, int hi) {
  v16h b;
#pragma unroll
  for (int i = 0; i < 16; ++i) b[i] = p[(i >> 3) * 16 + hi * 8 + (i & 7)];   // K = k0 + (i/8)*16 + hi*8 + i%8
  return b;
}

// ---------------- init: zero barrier counter + loss slots, hxs -> f16 ----------------
__global__ void k_init(const float* __restrict__ hxs, _Float16* __restrict__ h16,
                       float* __restrict__ out, int* __restrict__ ctr) {
  int i = blockIdx.x * 256 + threadIdx.x;
  if (i == 0) { *ctr = 0; out[OUT_AR] = 0.0f; out[OUT_TAR] = 0.0f; }
  if (i < B_ * H_) h16[i] = (_Float16)hxs[i];
}

// ---------------- generic f32 -> f16 conversion (4 elems/thread) ----------------
__global__ void k_cvt(const float* __restrict__ in, _Float16* __restrict__ out, int n4) {
  for (int i = blockIdx.x * 256 + threadIdx.x; i < n4; i += gridDim.x * 256) {
    float4 v = ((const float4*)in)[i];
    v4h h; h[0] = (_Float16)v.x; h[1] = (_Float16)v.y; h[2] = (_Float16)v.z; h[3] = (_Float16)v.w;
    ((v4h*)out)[i] = h;
  }
}

// ---------------- build padded head weight block: [actor(18); critic(1); zeros(13)] x 256 ----------------
__global__ void k_cvt_head(const float* __restrict__ actor_w, const float* __restrict__ critic_w,
                           _Float16* __restrict__ wpad) {
  int i = blockIdx.x * 256 + threadIdx.x;       // 32*256 elements
  if (i >= 32 * H_) return;
  int n = i >> 8, k = i & 255;
  float v = 0.0f;
  if (n < 18)       v = actor_w[(size_t)n * H_ + k];
  else if (n == 18) v = critic_w[k];
  wpad[i] = (_Float16)v;
}

// ---------------- GEMM core: block = 8 row-tiles x one 64-col group ----------------
// W slice for the column group is staged once in LDS and shared by all 8 waves;
// 4 independent accumulators per wave (no WMMA->WMMA RAW), A loaded once per K-chunk.
template <int KD>
__device__ __forceinline__ void gemm_block(const _Float16* __restrict__ A,
                                           const _Float16* __restrict__ W,
                                           _Float16* __restrict__ wsl,   // LDS, 64*(KD+8)
                                           int rowB, int cg, v8f (&acc)[4]) {
  constexpr int LDP = KD + 8;          // padded f16 row stride (spreads LDS banks)
  const int tid = threadIdx.x, lane = tid & 31, wave = tid >> 5;
  const int lo = lane & 15, hi = lane >> 4;
  // cooperative stage: 64 rows x KD f16, 8-byte vector copies
  const _Float16* Wb = W + (size_t)(cg * 64) * KD;
  for (int i = tid; i < 64 * KD / 4; i += 256) {
    int e = i * 4, nl = e / KD, k = e % KD;
    *(v4h*)&wsl[nl * LDP + k] = *(const v4h*)&Wb[(size_t)nl * KD + k];
  }
  __syncthreads();
  const int m0 = (rowB * 8 + wave) * 16;
  const _Float16* arow = A + (size_t)(m0 + lo) * KD;
#pragma unroll
  for (int kc = 0; kc < KD / 32; ++kc) {
    v16h a = load_a_f16(arow + kc * 32, hi);
#pragma unroll
    for (int j = 0; j < 4; ++j) {
      v16h bb = load_b_f16(&wsl[(j * 16 + lo) * LDP + kc * 32], hi);
      acc[j] = WMMA_F16(a, bb, acc[j]);
    }
  }
}

// ---------------- encoder layers: silu(A @ W^T + b) -> f16 ----------------
template <int KD, int NCG>     // KD = K dim, NCG = number of 64-col groups
__global__ void k_mlp(const _Float16* __restrict__ A, const _Float16* __restrict__ W,
                      const float* __restrict__ bias, _Float16* __restrict__ out, int ldo) {
  __shared__ _Float16 wsl[64 * (KD + 8)];
  const int rowB = blockIdx.x / NCG, cg = blockIdx.x % NCG;
  const int lane = threadIdx.x & 31, wave = threadIdx.x >> 5;
  const int lo = lane & 15, hi = lane >> 4;
  v8f acc[4] = {};
  gemm_block<KD>(A, W, wsl, rowB, cg, acc);
  const int m0 = (rowB * 8 + wave) * 16;
#pragma unroll
  for (int j = 0; j < 4; ++j) {
    int n = cg * 64 + j * 16 + lo;
    float bz = bias[n];
#pragma unroll
    for (int r = 0; r < 8; ++r) {
      int m = m0 + r + 8 * hi;
      float x = acc[j][r] + bz;
      out[(size_t)m * ldo + n] = (_Float16)(x * sigmoidf_(x));
    }
  }
}

// ---------------- x-gates: enc2 @ w_ih^T + (b_ih+b_hh), time-major f16 ----------------
__global__ void k_xgates(const _Float16* __restrict__ enc2, const _Float16* __restrict__ wih,
                         const float* __restrict__ b_ih, const float* __restrict__ b_hh,
                         _Float16* __restrict__ xg) {
  __shared__ _Float16 wsl[64 * (E_ + 8)];
  const int rowB = blockIdx.x >> 4, cg = blockIdx.x & 15;   // 512 x 16
  const int lane = threadIdx.x & 31, wave = threadIdx.x >> 5;
  const int lo = lane & 15, hi = lane >> 4;
  v8f acc[4] = {};
  gemm_block<E_>(enc2, wih, wsl, rowB, cg, acc);
  const int m0 = (rowB * 8 + wave) * 16;
#pragma unroll
  for (int j = 0; j < 4; ++j) {
    int n = cg * 64 + j * 16 + lo;
    float bz = b_ih[n] + b_hh[n];
#pragma unroll
    for (int r = 0; r < 8; ++r) {
      int m = m0 + r + 8 * hi;            // m = b*T + t
      int b = m >> 8, t = m & 255;
      xg[((size_t)t * B_ + b) * G4H + n] = (_Float16)(acc[j][r] + bz);
    }
  }
}

// ---------------- persistent LSTM recurrence: 16 WGs, grid barrier per step ----------------
#define LDW 264   // padded LDS row stride (f16) to spread banks

__global__ void k_lstm(const _Float16* __restrict__ whh, const float* __restrict__ cxs,
                       const _Float16* __restrict__ xg, _Float16* __restrict__ h16,
                       _Float16* __restrict__ hs16, float* __restrict__ out,
                       int* __restrict__ ctr) {
  __shared__ _Float16 wsl[64 * LDW];
  const int tid = threadIdx.x, lane = tid & 31, wave = tid >> 5;
  const int lo = lane & 15, hi = lane >> 4;
  const int c0 = blockIdx.x * 16;          // owned H-column slice [c0, c0+16)

  // stage w_hh slice (4 gates x 16 cols x 256 K) f16 -> LDS
  for (int i = tid; i < 64 * 256; i += 256) {
    int nl = i >> 8, k = i & 255;
    int n = ((nl >> 4) * H_) + c0 + (nl & 15);
    wsl[nl * LDW + k] = whh[(size_t)n * H_ + k];
  }
  __syncthreads();

  // persistent c-state in registers: wave owns row tiles {2w, 2w+1}
  const int rt0 = wave * 2;
  v8f cst[2];
#pragma unroll
  for (int q = 0; q < 2; ++q)
#pragma unroll
    for (int r = 0; r < 8; ++r) {
      int b = (rt0 + q) * 16 + r + 8 * hi;
      cst[q][r] = cxs[(size_t)b * H_ + c0 + lo];
    }

  const int h = c0 + lo;
  for (int t = 0; t < T_; ++t) {
    const _Float16* hcur = h16 + (size_t)(t & 1) * (B_ * H_);
    _Float16* hnxt = h16 + (size_t)((t + 1) & 1) * (B_ * H_);

#pragma unroll
    for (int q = 0; q < 2; ++q) {
      const int rt = rt0 + q;
      if (t + 1 < T_)   // warm L2 for next step's x-gates
        __builtin_prefetch(&xg[((size_t)(t + 1) * B_ + rt * 16 + lo) * G4H + c0], 0, 1);

      // accumulators seeded with precomputed x-gate contribution
      v8f acc[4];
#pragma unroll
      for (int gg = 0; gg < 4; ++gg) {
        int n = gg * H_ + c0 + lo;
#pragma unroll
        for (int r = 0; r < 8; ++r) {
          int b = rt * 16 + r + 8 * hi;
          acc[gg][r] = (float)xg[((size_t)t * B_ + b) * G4H + n];
        }
      }
      // gates += h_prev @ w_hh^T  (K = 256, 8 WMMA per gate tile, 4 independent accs)
      const _Float16* arow = hcur + (size_t)(rt * 16 + lo) * H_;
#pragma unroll
      for (int kc = 0; kc < 8; ++kc) {
        v16h a = load_a_f16(arow + kc * 32, hi);
#pragma unroll
        for (int gg = 0; gg < 4; ++gg) {
          v16h bb = load_b_f16(&wsl[(gg * 16 + lo) * LDW + kc * 32], hi);
          acc[gg] = WMMA_F16(a, bb, acc[gg]);
        }
      }
      // nonlinearities, cell update, fan-out stores
#pragma unroll
      for (int r = 0; r < 8; ++r) {
        int b = rt * 16 + r + 8 * hi;
        float iv = sigmoidf_(acc[0][r]);
        float fv = sigmoidf_(acc[1][r]);
        float gv = tanhf_(acc[2][r]);
        float ov = sigmoidf_(acc[3][r]);
        float c2 = fv * cst[q][r] + iv * gv;
        float h2 = ov * tanhf_(c2);
        cst[q][r] = c2;
        size_t base = ((size_t)b * T_ + t) * H_ + h;
        out[OUT_I + base]  = iv;
        out[OUT_F + base]  = fv;
        out[OUT_G + base]  = gv;
        out[OUT_O + base]  = ov;
        out[OUT_CS + base] = c2;
        out[OUT_HS + base] = h2;
        hnxt[(size_t)b * H_ + h] = (_Float16)h2;
        hs16[((size_t)b * T_ + t) * H_ + h] = (_Float16)h2;
        if (t == T_ - 1) {
          out[OUT_HT + (size_t)b * H_ + h] = h2;
          out[OUT_CT + (size_t)b * H_ + h] = c2;
        }
      }
    }
    // device-scope grid barrier across the 16 persistent workgroups
    __threadfence();
    __syncthreads();
    if (tid == 0) {
      __hip_atomic_fetch_add(ctr, 1, __ATOMIC_ACQ_REL, __HIP_MEMORY_SCOPE_AGENT);
      const int target = NWG * (t + 1);
      while (__hip_atomic_load(ctr, __ATOMIC_ACQUIRE, __HIP_MEMORY_SCOPE_AGENT) < target)
        __builtin_amdgcn_s_sleep(1);
    }
    __syncthreads();
  }
}

// ---------------- actor/critic heads: hs16 @ wpad^T (32 padded cols) ----------------
__global__ void k_head(const _Float16* __restrict__ hs16, const _Float16* __restrict__ wpad,
                       const float* __restrict__ actor_b, const float* __restrict__ critic_b,
                       float* __restrict__ out) {
  const int lane = threadIdx.x & 31, wave = threadIdx.x >> 5;
  const int lo = lane & 15, hi = lane >> 4;
  const int rowT = blockIdx.x * 8 + wave;      // 4096 row tiles, 2 col tiles each
  const int m0 = rowT * 16;
  const _Float16* arow = hs16 + (size_t)(m0 + lo) * H_;
  v8f acc[2] = {};
#pragma unroll
  for (int kc = 0; kc < 8; ++kc) {
    v16h a = load_a_f16(arow + kc * 32, hi);
#pragma unroll
    for (int j = 0; j < 2; ++j) {
      const _Float16* brow = wpad + (size_t)(j * 16 + lo) * H_ + kc * 32;
      acc[j] = WMMA_F16(a, load_b_f16(brow, hi), acc[j]);
    }
  }
#pragma unroll
  for (int j = 0; j < 2; ++j) {
    int n = j * 16 + lo;
    float bz = (n < 18) ? actor_b[n] : ((n == 18) ? critic_b[0] : 0.0f);
#pragma unroll
    for (int r = 0; r < 8; ++r) {
      int m = m0 + r + 8 * hi;           // m = b*T + t
      float x = acc[j][r] + bz;
      if (n < 18)       out[OUT_LOGITS + (size_t)m * A_ + n] = x;
      else if (n == 18) out[OUT_VALUES + (size_t)m] = x;
    }
  }
}

// ---------------- ar/tar losses over hs (already in d_out) ----------------
__global__ void k_loss(const float* __restrict__ hs, float* __restrict__ out) {
  __shared__ float sa[256], st[256];
  const size_t N = (size_t)BT * H_;
  float la = 0.0f, lt = 0.0f;
  for (size_t i = (size_t)blockIdx.x * 256 + threadIdx.x; i < N; i += (size_t)gridDim.x * 256) {
    float x = hs[i];
    la += x * x;
    size_t t = (i / H_) & (T_ - 1);    // layout: ((b*T + t)*H + h)
    if (t > 0) { float d = x - hs[i - H_]; lt += d * d; }
  }
  sa[threadIdx.x] = la; st[threadIdx.x] = lt;
  __syncthreads();
  for (int s = 128; s > 0; s >>= 1) {
    if (threadIdx.x < s) { sa[threadIdx.x] += sa[threadIdx.x + s]; st[threadIdx.x] += st[threadIdx.x + s]; }
    __syncthreads();
  }
  if (threadIdx.x == 0) {
    atomicAdd(out + OUT_AR,  sa[0] * (0.01f / (float)((size_t)BT * H_)));
    atomicAdd(out + OUT_TAR, st[0] * (0.01f / (float)((size_t)B_ * (T_ - 1) * H_)));
  }
}

// ---------------- host launcher ----------------
extern "C" void kernel_launch(void* const* d_in, const int* in_sizes, int n_in,
                              void* d_out, int out_size, void* d_ws, size_t ws_size,
                              hipStream_t stream) {
  (void)in_sizes; (void)n_in; (void)out_size; (void)ws_size;
  const float* obs      = (const float*)d_in[0];
  const float* hxs      = (const float*)d_in[1];
  const float* cxs      = (const float*)d_in[2];
  const float* enc_w1   = (const float*)d_in[3];
  const float* enc_b1   = (const float*)d_in[4];
  const float* enc_w2   = (const float*)d_in[5];
  const float* enc_b2   = (const float*)d_in[6];
  const float* w_ih     = (const float*)d_in[7];
  const float* w_hh     = (const float*)d_in[8];
  const float* b_ih     = (const float*)d_in[9];
  const float* b_hh     = (const float*)d_in[10];
  const float* actor_w  = (const float*)d_in[11];
  const float* actor_b  = (const float*)d_in[12];
  const float* critic_w = (const float*)d_in[13];
  const float* critic_b = (const float*)d_in[14];
  float* out = (float*)d_out;

  char* ws = (char*)d_ws;
  int*       ctr   = (int*)(ws + WS_CTR);
  _Float16*  h16   = (_Float16*)(ws + WS_H16);
  _Float16*  hs16  = (_Float16*)(ws + WS_HS16);
  _Float16*  enc1  = (_Float16*)(ws + WS_ENC1);
  _Float16*  enc2  = (_Float16*)(ws + WS_ENC2);
  _Float16*  xg    = (_Float16*)(ws + WS_XG);
  _Float16*  obs16 = (_Float16*)(ws + WS_OBS16);
  _Float16*  w1h   = (_Float16*)(ws + WS_W1);
  _Float16*  w2h   = (_Float16*)(ws + WS_W2);
  _Float16*  wihh  = (_Float16*)(ws + WS_WIH);
  _Float16*  whhh  = (_Float16*)(ws + WS_WHH);
  _Float16*  wpad  = (_Float16*)(ws + WS_WPAD);

  k_init<<<256, 256, 0, stream>>>(hxs, h16, out, ctr);

  // one-time f32 -> f16 staging (bandwidth-bound, tiny)
  k_cvt<<<2048, 256, 0, stream>>>(obs,    obs16, BT * D_ / 4);
  k_cvt<<<32,   256, 0, stream>>>(enc_w1, w1h,   E_ * D_ / 4);
  k_cvt<<<64,   256, 0, stream>>>(enc_w2, w2h,   E_ * E_ / 4);
  k_cvt<<<256,  256, 0, stream>>>(w_ih,   wihh,  G4H * E_ / 4);
  k_cvt<<<256,  256, 0, stream>>>(w_hh,   whhh,  G4H * H_ / 4);
  k_cvt_head<<<32, 256, 0, stream>>>(actor_w, critic_w, wpad);

  // encoder (silu MLP x2), input-gate GEMM, recurrence, heads, losses
  k_mlp<D_, 4><<<2048, 256, 0, stream>>>(obs16, w1h, enc_b1, enc1, E_);
  k_mlp<E_, 4><<<2048, 256, 0, stream>>>(enc1,  w2h, enc_b2, enc2, E_);
  k_xgates<<<8192, 256, 0, stream>>>(enc2, wihh, b_ih, b_hh, xg);
  k_lstm<<<NWG, 256, 0, stream>>>(whhh, cxs, xg, h16, hs16, out, ctr);
  k_head<<<512, 256, 0, stream>>>(hs16, wpad, actor_b, critic_b, out);
  k_loss<<<2048, 256, 0, stream>>>(out + OUT_HS, out);
}
